// BiLSTMAnomalyDetectionModel_45002667327857
// MI455X (gfx1250) — compile-verified, matrix operand order checked by reading, not run on hardware
//
#include <hip/hip_runtime.h>
#include <hip/hip_bf16.h>

typedef __attribute__((ext_vector_type(16))) _Float16     v16h;
typedef __attribute__((ext_vector_type(8)))  float        v8f;
typedef __attribute__((ext_vector_type(4)))  _Float16     v4h;
typedef __attribute__((ext_vector_type(4)))  unsigned int v4u;
typedef __attribute__((ext_vector_type(8)))  int          v8i;
typedef __attribute__((ext_vector_type(4)))  int          v4i;

#define Bsz 256
#define Tsz 1024
#define Fsz 128
#define G4H 256   // 4*H
#define Hsz 64

__device__ __forceinline__ float sigmoidf_(float x) {
    return 1.f / (1.f + __expf(-x));
}

// Issue one TDM copy of a contiguous 16KB tile (2048 x 8B) global -> LDS.
// D# group0/group1 packed per CDNA5 ISA section 8.3/8.4 (1-D tensor).
// clang-23 toolchain: 6-arg builtin (g0, g1, g2, g3, g4, cpol).
__device__ __forceinline__ void tdm_load_16kb(const _Float16* gsrc, unsigned lds_off) {
    unsigned long long ga = (unsigned long long)gsrc;
    v4u g0;
    g0[0] = 1u;                                            // count=1, user mode
    g0[1] = lds_off;                                       // lds_addr
    g0[2] = (unsigned)ga;                                  // global_addr[31:0]
    g0[3] = ((unsigned)(ga >> 32) & 0x01FFFFFFu)           // global_addr[56:32]
            | 0x80000000u;                                 // type=2 ("image")
    v8i g1;
    g1[0] = 0x00030000;                                    // data_size=3 (8B), no multicast
    g1[1] = (int)(2048u << 16);                            // tensor_dim0 = 2048
    g1[2] = (int)(1u << 16);                               // tensor_dim1 = 1
    g1[3] = (int)(2048u << 16);                            // tile_dim0 = 2048
    g1[4] = 0;                                             // tile_dim1/2 unused
    g1[5] = 2048;                                          // tensor_dim0_stride
    g1[6] = 0;
    g1[7] = 0;
    v4i g2 = {0, 0, 0, 0};
    v4i g3 = {0, 0, 0, 0};
    v8i g4 = {0, 0, 0, 0, 0, 0, 0, 0};
    __builtin_amdgcn_tensor_load_to_lds(g0, g1, g2, g3, g4, 0);
}

// ---------------------------------------------------------------------------
// Kernel A: input projection  xg[t,b,g] = x[b,t,:] @ kernel[:,g] + bias[g]
// One block per (t, dir). 8 waves; wave w covers batch rows [32w, 32w+32).
// ---------------------------------------------------------------------------
__global__ void __launch_bounds__(256) lstm_inproj_kernel(
    const float* __restrict__ x,
    const float* __restrict__ kernel_f, const float* __restrict__ bias_f,
    const float* __restrict__ kernel_b, const float* __restrict__ bias_b,
    _Float16* __restrict__ xg_f, _Float16* __restrict__ xg_b)
{
    extern __shared__ _Float16 klds[];          // [256][132] = [g][f], padded
    const int t   = blockIdx.x;
    const int dir = blockIdx.y;
    const float* W   = dir ? kernel_b : kernel_f;
    const float* bia = dir ? bias_b   : bias_f;
    _Float16*    xg  = dir ? xg_b     : xg_f;

    const int tid = threadIdx.x;
    for (int i = 0; i < Fsz; ++i)
        klds[tid * 132 + i] = (_Float16)W[i * G4H + tid];
    __syncthreads();

    const int wv   = tid >> 5;
    const int lane = tid & 31;
    const int m    = lane & 15;
    const int half = lane >> 4;

    // Hoist bias loads out of the tile loop (one per column tile).
    float barr[16];
    #pragma unroll
    for (int ct = 0; ct < 16; ++ct) barr[ct] = bia[ct * 16 + m];

    // A fragments: 2 row tiles x 4 K-fragments, f32 -> f16 on the fly.
    v16h afrag[2][4];
    for (int rt = 0; rt < 2; ++rt) {
        const int brow = (wv * 2 + rt) * 16 + m;
        const float* xrow = x + ((size_t)brow * Tsz + t) * Fsz;
        for (int kf = 0; kf < 4; ++kf) {
            const int k0 = kf * 32 + half * 8;
            union { v16h v; _Float16 h[16]; } u;
            #pragma unroll
            for (int j = 0; j < 8; ++j) {
                u.h[j]     = (_Float16)xrow[k0 + j];
                u.h[j + 8] = (_Float16)xrow[k0 + 16 + j];
            }
            afrag[rt][kf] = u.v;
        }
    }

    for (int ct = 0; ct < 16; ++ct) {
        v16h bfrag[4];
        const int gcol = ct * 16 + m;
        #pragma unroll
        for (int kf = 0; kf < 4; ++kf) {
            const _Float16* bp = klds + gcol * 132 + kf * 32 + half * 8;
            union { v16h v; v4h q[4]; } u;
            u.q[0] = *(const v4h*)(bp + 0);
            u.q[1] = *(const v4h*)(bp + 4);
            u.q[2] = *(const v4h*)(bp + 16);
            u.q[3] = *(const v4h*)(bp + 20);
            bfrag[kf] = u.v;
        }
        for (int rt = 0; rt < 2; ++rt) {
            union { v8f v; float f[8]; } c;
            #pragma unroll
            for (int r = 0; r < 8; ++r) c.f[r] = barr[ct];
            #pragma unroll
            for (int kf = 0; kf < 4; ++kf)
                c.v = __builtin_amdgcn_wmma_f32_16x16x32_f16(
                        false, afrag[rt][kf], false, bfrag[kf],
                        (short)0, c.v, false, false);
            const size_t base = (size_t)t * Bsz * G4H;
            #pragma unroll
            for (int r = 0; r < 8; ++r) {
                const int brow = (wv * 2 + rt) * 16 + r + half * 8;
                xg[base + (size_t)brow * G4H + ct * 16 + m] = (_Float16)c.f[r];
            }
        }
    }
}

// ---------------------------------------------------------------------------
// Kernel B: recurrence. One block per (batch-tile of 32, dir).
// Wave w owns gate-group j=w>>1 and row-tile rt=w&1: its 4 WMMA accumulators
// hold i/f/g/o for the same (row, h-col) pairs -> gate math fully in regs.
// xg tiles streamed by the Tensor Data Mover into a 4-deep LDS ring,
// prefetched 2 steps ahead; h state double-buffered in LDS.
// ONE barrier per serial step.
// ---------------------------------------------------------------------------
__global__ void __launch_bounds__(256) lstm_recur_kernel(
    const float* __restrict__ rec_f, const float* __restrict__ rec_b,
    const _Float16* __restrict__ xg_f, const _Float16* __restrict__ xg_b,
    _Float16* __restrict__ h_f, _Float16* __restrict__ h_b)
{
    __shared__ __align__(16) _Float16 xbuf[4][32 * G4H];  // 4 x 16KB TDM ring
    __shared__ __align__(16) _Float16 hbuf[2][32 * 68];   // double-buffered h

    const int dir = blockIdx.y;
    const float*    rec  = dir ? rec_b : rec_f;
    const _Float16* xg   = dir ? xg_b  : xg_f;
    _Float16*       hout = dir ? h_b   : h_f;
    const int bBase = blockIdx.x * 32;

    const int tid  = threadIdx.x;
    const int wv   = tid >> 5;
    const int lane = tid & 31;
    const int m    = lane & 15;
    const int half = lane >> 4;
    const int jg   = wv >> 1;      // gate-column group (h cols [16j,16j+16))
    const int rt   = wv & 1;       // row tile (batch rows [16rt,16rt+16))

    // Recurrent-weight B fragments: 4 gate tiles x 2 K-fragments (resident).
    v16h bfrag[4][2];
    for (int g = 0; g < 4; ++g) {
        const int col = (jg + 4 * g) * 16 + m;
        for (int kf = 0; kf < 2; ++kf) {
            const int k0 = kf * 32 + half * 8;
            union { v16h v; _Float16 h[16]; } u;
            #pragma unroll
            for (int j = 0; j < 8; ++j) {
                u.h[j]     = (_Float16)rec[(size_t)(k0 + j)      * G4H + col];
                u.h[j + 8] = (_Float16)rec[(size_t)(k0 + 16 + j) * G4H + col];
            }
            bfrag[g][kf] = u.v;
        }
    }

    unsigned lds_x[4];
    #pragma unroll
    for (int i = 0; i < 4; ++i)
        lds_x[i] = (unsigned)(unsigned long long)(&xbuf[i][0]);

    // Prologue: start TDM copies of the first two tiles; zero h state.
    if (wv == 0) {
        const int t0 = dir ? (Tsz - 1) : 0;
        const int t1 = dir ? (Tsz - 2) : 1;
        tdm_load_16kb(xg + ((size_t)t0 * Bsz + bBase) * G4H, lds_x[0]);
        tdm_load_16kb(xg + ((size_t)t1 * Bsz + bBase) * G4H, lds_x[1]);
    }
    for (int i = tid; i < 32 * 68; i += 256) hbuf[0][i] = (_Float16)0.f;
    float cst[8];
    #pragma unroll
    for (int r = 0; r < 8; ++r) cst[r] = 0.f;

    int rb = 0;   // ring buffer slot read at step s   (s   % 4)
    int ib = 2;   // ring buffer slot filled at step s (s+2 % 4)
    for (int s = 0; s < Tsz; ++s) {
        const int t = dir ? (Tsz - 1 - s) : s;
        if (wv == 0) {
            if (s + 2 < Tsz) {   // prefetch 2 steps ahead, then wait current
                const int tn = dir ? (Tsz - 3 - s) : (s + 2);
                tdm_load_16kb(xg + ((size_t)tn * Bsz + bBase) * G4H, lds_x[ib]);
                __builtin_amdgcn_s_wait_tensorcnt((short)2);
            } else if (s + 1 < Tsz) {
                __builtin_amdgcn_s_wait_tensorcnt((short)1);
            } else {
                __builtin_amdgcn_s_wait_tensorcnt((short)0);
            }
        }
        __syncthreads();   // single barrier: xbuf[rb] + hbuf[s&1] published

        // A fragments (h state) from LDS.
        v16h afrag[2];
        {
            const _Float16* hp = &hbuf[s & 1][0] + (rt * 16 + m) * 68;
            #pragma unroll
            for (int kf = 0; kf < 2; ++kf) {
                const int k0 = kf * 32 + half * 8;
                union { v16h v; v4h q[4]; } u;
                u.q[0] = *(const v4h*)(hp + k0);
                u.q[1] = *(const v4h*)(hp + k0 + 4);
                u.q[2] = *(const v4h*)(hp + k0 + 16);
                u.q[3] = *(const v4h*)(hp + k0 + 20);
                afrag[kf] = u.v;
            }
        }
        // z = xg_t + h @ rec : 4 gate tiles, xg read from TDM-staged LDS.
        const _Float16* xs = &xbuf[rb][0];
        union { v8f v; float f[8]; } cacc[4];
        #pragma unroll
        for (int g = 0; g < 4; ++g) {
            const int colb = (jg + 4 * g) * 16;
            #pragma unroll
            for (int r = 0; r < 8; ++r) {
                const int row = rt * 16 + r + half * 8;
                cacc[g].f[r] = (float)xs[row * G4H + colb + m];
            }
            cacc[g].v = __builtin_amdgcn_wmma_f32_16x16x32_f16(
                    false, afrag[0], false, bfrag[g][0], (short)0, cacc[g].v, false, false);
            cacc[g].v = __builtin_amdgcn_wmma_f32_16x16x32_f16(
                    false, afrag[1], false, bfrag[g][1], (short)0, cacc[g].v, false, false);
        }

        // Gate math fully in registers: lane owns (row, h-col) for r=0..7.
        _Float16* hw = &hbuf[(s + 1) & 1][0];
        const int colh = jg * 16 + m;
        #pragma unroll
        for (int r = 0; r < 8; ++r) {
            const int row = rt * 16 + r + half * 8;
            const float ig = sigmoidf_(cacc[0].f[r]);
            const float fg = sigmoidf_(cacc[1].f[r]);
            const float gg = tanhf(cacc[2].f[r]);
            const float og = sigmoidf_(cacc[3].f[r]);
            const float cv = fg * cst[r] + ig * gg;
            cst[r] = cv;
            const float hv = og * tanhf(cv);
            hw[row * 68 + colh] = (_Float16)hv;
            hout[((size_t)t * Bsz + bBase + row) * Hsz + colh] = (_Float16)hv;
        }
        rb = (rb + 1) & 3;
        ib = (ib + 1) & 3;
    }
}

// ---------------------------------------------------------------------------
// Kernel C: out[b,t] = sigmoid([h_f ; h_b] . dense_w + dense_b)
// ---------------------------------------------------------------------------
__global__ void __launch_bounds__(256) lstm_dense_kernel(
    const _Float16* __restrict__ h_f, const _Float16* __restrict__ h_b,
    const float* __restrict__ dense_w, const float* __restrict__ dense_b,
    float* __restrict__ out)
{
    __shared__ float wsh[2 * Hsz];
    const int t = blockIdx.x;
    const int b = threadIdx.x;
    if (b < 2 * Hsz) wsh[b] = dense_w[b];
    __syncthreads();
    const _Float16* pf = h_f + ((size_t)t * Bsz + b) * Hsz;
    const _Float16* pb = h_b + ((size_t)t * Bsz + b) * Hsz;
    float acc = dense_b[0];
    #pragma unroll
    for (int j = 0; j < Hsz; ++j) acc += (float)pf[j] * wsh[j];
    #pragma unroll
    for (int j = 0; j < Hsz; ++j) acc += (float)pb[j] * wsh[Hsz + j];
    out[(size_t)b * Tsz + t] = sigmoidf_(acc);
}

extern "C" void kernel_launch(void* const* d_in, const int* in_sizes, int n_in,
                              void* d_out, int out_size, void* d_ws, size_t ws_size,
                              hipStream_t stream) {
    (void)in_sizes; (void)n_in; (void)out_size; (void)ws_size;
    const float* x        = (const float*)d_in[0];
    const float* kernel_f = (const float*)d_in[1];
    const float* rec_f    = (const float*)d_in[2];
    const float* bias_f   = (const float*)d_in[3];
    const float* kernel_b = (const float*)d_in[4];
    const float* rec_b    = (const float*)d_in[5];
    const float* bias_b   = (const float*)d_in[6];
    const float* dense_w  = (const float*)d_in[7];
    const float* dense_b  = (const float*)d_in[8];
    float* out = (float*)d_out;

    char* ws = (char*)d_ws;
    // Workspace: xg_f/xg_b [T,B,4H] f16 (128MB each), h_f/h_b [T,B,H] f16 (32MB each)
    _Float16* xg_f = (_Float16*)(ws);
    _Float16* xg_b = (_Float16*)(ws + 134217728u);
    _Float16* h_f  = (_Float16*)(ws + 268435456u);
    _Float16* h_b  = (_Float16*)(ws + 301989888u);

    lstm_inproj_kernel<<<dim3(Tsz, 2), 256, 256 * 132 * sizeof(_Float16), stream>>>(
        x, kernel_f, bias_f, kernel_b, bias_b, xg_f, xg_b);
    lstm_recur_kernel<<<dim3(Bsz / 32, 2), 256, 0, stream>>>(
        rec_f, rec_b, xg_f, xg_b, h_f, h_b);
    lstm_dense_kernel<<<Tsz, 256, 0, stream>>>(h_f, h_b, dense_w, dense_b, out);
}
